// SikeSlabLinear_46402826666275
// MI455X (gfx1250) — compile-verified, matrix operand order checked by reading, not run on hardware
//
#include <hip/hip_runtime.h>
#include <hip/hip_bf16.h>

#define B_DIM 4096
#define K_DIM 1024
#define L_DIM 16384
#define O_DIM 2048

typedef __attribute__((ext_vector_type(16))) __bf16 v16bf;
typedef __attribute__((ext_vector_type(8)))  __bf16 v8bf;
typedef __attribute__((ext_vector_type(8)))  float  v8f;
typedef unsigned short ushort_t;
typedef unsigned int   uint_t;
typedef unsigned long long u64_t;

// ---------- helpers ----------
__device__ __forceinline__ ushort_t f2bf(float f) {
  uint_t u = __builtin_bit_cast(uint_t, f);
  u = u + 0x7FFFu + ((u >> 16) & 1u);        // round-to-nearest-even
  return (ushort_t)(u >> 16);
}

__device__ __forceinline__ uint_t rotl32(uint_t x, int r) {
  return (x << r) | (x >> (32 - r));
}

// threefry-2x32, 20 rounds (JAX PRNG), key = (0, 42)
__device__ __forceinline__ void threefry2x32(uint_t k0, uint_t k1,
                                             uint_t& x0, uint_t& x1) {
  uint_t ks[3] = {k0, k1, 0x1BD11BDAu ^ k0 ^ k1};
  const int RA[4] = {13, 15, 26, 6};
  const int RB[4] = {17, 29, 16, 24};
  x0 += ks[0]; x1 += ks[1];
  #pragma unroll
  for (int g = 0; g < 5; ++g) {
    const int* rot = (g & 1) ? RB : RA;
    #pragma unroll
    for (int r = 0; r < 4; ++r) { x0 += x1; x1 = rotl32(x1, rot[r]); x1 ^= x0; }
    x0 += ks[(g + 1) % 3];
    x1 += ks[(g + 2) % 3] + (uint_t)(g + 1);
  }
}

// monotonic float -> orderable uint
__device__ __forceinline__ uint_t f32_orderable(float f) {
  uint_t b = __builtin_bit_cast(uint_t, f);
  return (b & 0x80000000u) ? ~b : (b | 0x80000000u);
}

// ISA 16-bit fragment: two contiguous 8-element K runs at h*8 and 16+h*8
__device__ __forceinline__ v16bf load_frag(const ushort_t* row, int k0, int h) {
  v8bf lo = *(const v8bf*)(row + k0 + h * 8);
  v8bf hi = *(const v8bf*)(row + k0 + 16 + h * 8);
  return __builtin_shufflevector(lo, hi, 0,1,2,3,4,5,6,7,8,9,10,11,12,13,14,15);
}

// ---------- phase 0: precision conversion ----------
__global__ void zero_s_kernel(float* s) {
  int i = blockIdx.x * blockDim.x + threadIdx.x;
  if (i < L_DIM) s[i] = 0.0f;
}

__global__ void convert_x_kernel(const float* __restrict__ x,
                                 ushort_t* __restrict__ xb, int n) {
  for (int i = blockIdx.x * blockDim.x + threadIdx.x; i < n;
       i += gridDim.x * blockDim.x)
    xb[i] = f2bf(x[i]);
}

// w[k][l] (f32, row-major 1024x16384) -> wt[l][k] (bf16, 16384x1024)
__global__ __launch_bounds__(256) void transpose_w_kernel(
    const float* __restrict__ w, ushort_t* __restrict__ wt) {
  __shared__ float tile[32][33];
  const int tx = threadIdx.x & 31;
  const int ty = threadIdx.x >> 5;        // 0..7
  const int l0 = blockIdx.x * 32;        // along L
  const int k0 = blockIdx.y * 32;        // along K
  #pragma unroll
  for (int i = 0; i < 4; ++i) {
    int k = k0 + ty + i * 8;
    tile[ty + i * 8][tx] = w[(size_t)k * L_DIM + l0 + tx];
  }
  __syncthreads();
  #pragma unroll
  for (int i = 0; i < 4; ++i) {
    int l = l0 + ty + i * 8;
    wt[(size_t)l * K_DIM + k0 + tx] = f2bf(tile[tx][ty + i * 8]);
  }
}

// ---------- phase 1: fused score GEMM: s[l] += sum_m |(x@w)[m,l]| ----------
// 8 waves = 2(M) x 4(N); wave tile = 64(M) x 32(N) via 4x2 register blocking.
// Block tile = 128(M) x 128(N). 8 WMMAs per 6 fragment loads.
__global__ __launch_bounds__(256) void score_gemm_kernel(
    const ushort_t* __restrict__ xb, const ushort_t* __restrict__ wt,
    float* __restrict__ s) {
  const int lane  = threadIdx.x & 31;
  const int wave  = threadIdx.x >> 5;
  const int waveM = wave >> 2;           // 0..1
  const int waveN = wave & 3;            // 0..3
  const int n     = lane & 15;           // fragment lane row/col
  const int h     = lane >> 4;           // half-wave selector
  const int mWave = blockIdx.y * 128 + waveM * 64;
  const int nWave = blockIdx.x * 128 + waveN * 32;

  const ushort_t* aRow[4];
  const ushort_t* bRow[2];
  #pragma unroll
  for (int i = 0; i < 4; ++i) aRow[i] = xb + (size_t)(mWave + i * 16 + n) * K_DIM;
  #pragma unroll
  for (int j = 0; j < 2; ++j) bRow[j] = wt + (size_t)(nWave + j * 16 + n) * K_DIM;

  v8f c[4][2] = {};
  #pragma unroll 2
  for (int k0 = 0; k0 < K_DIM; k0 += 32) {
    v16bf a[4], b[2];
    #pragma unroll
    for (int i = 0; i < 4; ++i) a[i] = load_frag(aRow[i], k0, h);
    #pragma unroll
    for (int j = 0; j < 2; ++j) b[j] = load_frag(bRow[j], k0, h);
    #pragma unroll
    for (int i = 0; i < 4; ++i)
      #pragma unroll
      for (int j = 0; j < 2; ++j)
        c[i][j] = __builtin_amdgcn_wmma_f32_16x16x32_bf16(
            false, a[i], false, b[j], (short)0, c[i][j], false, false);
  }

  // per lane: column N = n (within each N-tile), rows M = r + 8*h of 4 M-tiles
  #pragma unroll
  for (int j = 0; j < 2; ++j) {
    float t = 0.0f;
    #pragma unroll
    for (int i = 0; i < 4; ++i)
      #pragma unroll
      for (int r = 0; r < 8; ++r) t += __builtin_fabsf(c[i][j][r]);
    t += __shfl_xor(t, 16, 32);          // merge half-columns (wave32)
    if (lane < 16) atomicAdd(&s[nWave + j * 16 + lane], t);
  }
}

// ---------- phase 2: max, gates, rank packing ----------
__global__ __launch_bounds__(256) void reduce_max_kernel(
    const float* __restrict__ s, float* __restrict__ smax) {
  __shared__ float red[256];
  float m = -1e30f;
  for (int i = threadIdx.x; i < L_DIM; i += 256) m = fmaxf(m, s[i]);
  red[threadIdx.x] = m;
  __syncthreads();
  for (int st = 128; st > 0; st >>= 1) {
    if (threadIdx.x < st) red[threadIdx.x] = fmaxf(red[threadIdx.x], red[threadIdx.x + st]);
    __syncthreads();
  }
  if (threadIdx.x == 0) *smax = red[0] + 1.0f;   // "big"
}

// pack sortable u64: high 32 = ~orderable(rank) (=> ascending sort == key desc),
// low 32 = idx (=> ascending idx tiebreak, matching jax top_k)
__global__ __launch_bounds__(256) void rank_pack_kernel(
    const float* __restrict__ s, const float* __restrict__ smax,
    const float* __restrict__ pi, u64_t* __restrict__ v) {
  int i = blockIdx.x * blockDim.x + threadIdx.x;   // 0..8191, pairs (i, i+8192)
  if (i >= L_DIM / 2) return;
  const float p   = 1.0f / (1.0f + __expf(-pi[0]));   // sigmoid(pi)
  const float big = *smax;
  uint_t x0 = (uint_t)i, x1 = (uint_t)(i + L_DIM / 2);
  threefry2x32(0u, 42u, x0, x1);                      // jax.random.key(42)
  float u0 = __builtin_bit_cast(float, (x0 >> 9) | 0x3f800000u) - 1.0f;
  float u1 = __builtin_bit_cast(float, (x1 >> 9) | 0x3f800000u) - 1.0f;
  float r0 = s[i]             + ((u0 < p) ? big : 0.0f);
  float r1 = s[i + L_DIM / 2] + ((u1 < p) ? big : 0.0f);
  v[i]             = ((u64_t)(~f32_orderable(r0)) << 32) | (uint_t)i;
  v[i + L_DIM / 2] = ((u64_t)(~f32_orderable(r1)) << 32) | (uint_t)(i + L_DIM / 2);
}

// ---------- phase 3: bitonic sort (ascending u64) ----------
__global__ __launch_bounds__(256) void bitonic_step_kernel(
    u64_t* __restrict__ v, int j, int k) {
  int i = blockIdx.x * blockDim.x + threadIdx.x;
  int ixj = i ^ j;
  if (ixj > i) {
    u64_t a = v[i], b = v[ixj];
    bool up = ((i & k) == 0);
    if ((a > b) == up) { v[i] = b; v[ixj] = a; }
  }
}

__global__ void emit_idx_kernel(const u64_t* __restrict__ v,
                                int* __restrict__ idxOut) {
  int j = blockIdx.x * blockDim.x + threadIdx.x;
  if (j < O_DIM) idxOut[j] = (int)(v[j] & 0xFFFFFFFFu);
}

// ---------- phase 4: gathered GEMM out = x @ w[:, idx] ----------
// Same 4x2 register-blocked structure; B rows gathered via idx.
__global__ __launch_bounds__(256) void out_gemm_kernel(
    const ushort_t* __restrict__ xb, const ushort_t* __restrict__ wt,
    const int* __restrict__ idx, float* __restrict__ out) {
  const int lane  = threadIdx.x & 31;
  const int wave  = threadIdx.x >> 5;
  const int waveM = wave >> 2;
  const int waveN = wave & 3;
  const int n     = lane & 15;
  const int h     = lane >> 4;
  const int mWave = blockIdx.y * 128 + waveM * 64;
  const int nWave = blockIdx.x * 128 + waveN * 32;

  const ushort_t* aRow[4];
  const ushort_t* bRow[2];
  #pragma unroll
  for (int i = 0; i < 4; ++i) aRow[i] = xb + (size_t)(mWave + i * 16 + n) * K_DIM;
  #pragma unroll
  for (int j = 0; j < 2; ++j)
    bRow[j] = wt + (size_t)idx[nWave + j * 16 + n] * K_DIM;  // column gather

  v8f c[4][2] = {};
  #pragma unroll 2
  for (int k0 = 0; k0 < K_DIM; k0 += 32) {
    v16bf a[4], b[2];
    #pragma unroll
    for (int i = 0; i < 4; ++i) a[i] = load_frag(aRow[i], k0, h);
    #pragma unroll
    for (int j = 0; j < 2; ++j) b[j] = load_frag(bRow[j], k0, h);
    #pragma unroll
    for (int i = 0; i < 4; ++i)
      #pragma unroll
      for (int j = 0; j < 2; ++j)
        c[i][j] = __builtin_amdgcn_wmma_f32_16x16x32_bf16(
            false, a[i], false, b[j], (short)0, c[i][j], false, false);
  }

  #pragma unroll
  for (int i = 0; i < 4; ++i)
    #pragma unroll
    for (int j = 0; j < 2; ++j)
      #pragma unroll
      for (int r = 0; r < 8; ++r)
        out[(size_t)(mWave + i * 16 + r + 8 * h) * O_DIM + nWave + j * 16 + n] =
            c[i][j][r];
}

// ---------- host ----------
extern "C" void kernel_launch(void* const* d_in, const int* in_sizes, int n_in,
                              void* d_out, int out_size, void* d_ws, size_t ws_size,
                              hipStream_t stream) {
  const float* x  = (const float*)d_in[0];   // [4096,1024]
  const float* w  = (const float*)d_in[1];   // [1024,16384]
  const float* pi = (const float*)d_in[2];   // scalar

  float* out    = (float*)d_out;                          // [4096,2048]
  int*   idxOut = (int*)((float*)d_out + (size_t)B_DIM * O_DIM); // [2048]

  char* ws = (char*)d_ws;
  ushort_t* xb   = (ushort_t*)ws;                               // 8 MB
  ushort_t* wt   = (ushort_t*)(ws + (size_t)B_DIM * K_DIM * 2); // 32 MB
  char* ws2      = ws + (size_t)B_DIM * K_DIM * 2 + (size_t)L_DIM * K_DIM * 2;
  float* s       = (float*)ws2;                                 // 64 KB
  float* smax    = (float*)(ws2 + L_DIM * 4);                   // 256 B
  u64_t* v       = (u64_t*)(ws2 + L_DIM * 4 + 256);             // 128 KB

  // phase 0
  zero_s_kernel<<<L_DIM / 256, 256, 0, stream>>>(s);
  convert_x_kernel<<<4096, 256, 0, stream>>>(x, xb, B_DIM * K_DIM);
  transpose_w_kernel<<<dim3(L_DIM / 32, K_DIM / 32), 256, 0, stream>>>(w, wt);

  // phase 1: fused score GEMM (4096 x 1024 x 16384, never materialized)
  score_gemm_kernel<<<dim3(L_DIM / 128, B_DIM / 128), 256, 0, stream>>>(xb, wt, s);

  // phase 2
  reduce_max_kernel<<<1, 256, 0, stream>>>(s, smax);
  rank_pack_kernel<<<(L_DIM / 2) / 256, 256, 0, stream>>>(s, smax, pi, v);

  // phase 3: bitonic sort of 16384 packed keys (105 tiny launches)
  for (int k = 2; k <= L_DIM; k <<= 1)
    for (int j = k >> 1; j >= 1; j >>= 1)
      bitonic_step_kernel<<<L_DIM / 256, 256, 0, stream>>>(v, j, k);
  emit_idx_kernel<<<O_DIM / 256, 256, 0, stream>>>(v, idxOut);

  // phase 4: gathered GEMM (4096 x 1024 x 2048)
  out_gemm_kernel<<<dim3(O_DIM / 128, B_DIM / 128), 256, 0, stream>>>(xb, wt, idxOut, out);
}
